// ourModel_85023172592601
// MI455X (gfx1250) — matrix operand order repaired and checked by reading, not compiled, not run on hardware
//
#include <hip/hip_runtime.h>

typedef __attribute__((ext_vector_type(16))) __bf16 v16bf;
typedef __attribute__((ext_vector_type(8)))  float  v8f;

#define N_NODES 50000
#define N_EDGES 400000
#define DIM     128
#define NCLASS  40
#define KB      4
#define BN_EPS  1e-5f

// ---------------------------------------------------------------------------
// Encoder: per-node label-routed relu(We[lab] @ h[n] + be[lab]) -> enc[N,4]
// ---------------------------------------------------------------------------
__global__ void enc_kernel(const float* __restrict__ h,
                           const int*   __restrict__ labels,
                           const float* __restrict__ We,   // [C,K,D]
                           const float* __restrict__ be,   // [C,K]
                           float*       __restrict__ enc)  // [N,4]
{
    int lane = threadIdx.x & 31;
    int node = blockIdx.x * 8 + (threadIdx.x >> 5);
    if (node >= N_NODES) return;
    int lab = labels[node];
    if (lab < 0) {
        if (lane < 4) enc[node * 4 + lane] = 0.0f;
        return;
    }
    const float* hrow = h + (size_t)node * DIM;
    const float* w    = We + (size_t)lab * KB * DIM;
    float a0 = 0.f, a1 = 0.f, a2 = 0.f, a3 = 0.f;
#pragma unroll
    for (int j = 0; j < 4; ++j) {
        int d = lane + 32 * j;
        float hv = hrow[d];
        a0 = fmaf(w[0 * DIM + d], hv, a0);
        a1 = fmaf(w[1 * DIM + d], hv, a1);
        a2 = fmaf(w[2 * DIM + d], hv, a2);
        a3 = fmaf(w[3 * DIM + d], hv, a3);
    }
#pragma unroll
    for (int off = 16; off > 0; off >>= 1) {
        a0 += __shfl_xor(a0, off, 32);
        a1 += __shfl_xor(a1, off, 32);
        a2 += __shfl_xor(a2, off, 32);
        a3 += __shfl_xor(a3, off, 32);
    }
    if (lane == 0) {
        const float* b = be + (size_t)lab * KB;
        enc[node * 4 + 0] = fmaxf(a0 + b[0], 0.f);
        enc[node * 4 + 1] = fmaxf(a1 + b[1], 0.f);
        enc[node * 4 + 2] = fmaxf(a2 + b[2], 0.f);
        enc[node * 4 + 3] = fmaxf(a3 + b[3], 0.f);
    }
}

// ---------------------------------------------------------------------------
// Edge decoder + masked scatter: one wave32 per edge.
// num[N,128] (25.6MB) is L2-resident (192MB), so atomics stay on-chip.
// ---------------------------------------------------------------------------
__global__ void edge_kernel(const int*   __restrict__ srcI,
                            const int*   __restrict__ dstI,
                            const float* __restrict__ conf,
                            const int*   __restrict__ labels,
                            const float* __restrict__ enc,   // [N,4]
                            const float* __restrict__ Wd,    // [C,D,K]
                            const float* __restrict__ bd,    // [C,D]
                            float*       __restrict__ num,   // [N,D]
                            float*       __restrict__ den)   // [N]
{
    int lane = threadIdx.x & 31;
    int e    = blockIdx.x * 8 + (threadIdx.x >> 5);
    if (e >= N_EDGES) return;
    if (!(conf[e] > 0.5f)) return;           // edge not selected
    int d = dstI[e];
    if (lane == 0) atomicAdd(&den[d], 1.0f); // mask counts even if label<0
    int lab = labels[d];
    if (lab < 0) return;                     // dec contributes zeros
    int s = srcI[e];
    const float4 m = *(const float4*)(enc + (size_t)s * 4);
    const float* w = Wd + (size_t)lab * DIM * KB;
    const float* b = bd + (size_t)lab * DIM;
    float* nrow = num + (size_t)d * DIM;
#pragma unroll
    for (int j = 0; j < 4; ++j) {
        int dd = lane + 32 * j;
        float4 wr = *(const float4*)(w + dd * 4);
        float v = fmaf(wr.x, m.x, fmaf(wr.y, m.y, fmaf(wr.z, m.z, wr.w * m.w))) + b[dd];
        v = fmaxf(v, 0.f);
        atomicAdd(&nrow[dd], v);
    }
}

// agg = num / max(den,1), in place
__global__ void finalize_kernel(float* __restrict__ num, const float* __restrict__ den)
{
    int i = blockIdx.x * 256 + threadIdx.x;
    if (i >= N_NODES * DIM) return;
    int n = i >> 7;
    num[i] *= __frcp_rn(fmaxf(den[n], 1.0f));
}

// f32 -> bf16 weight packing (once per call)
__global__ void pack_bf16_kernel(const float* __restrict__ in,
                                 __bf16* __restrict__ out, int n)
{
    int i = blockIdx.x * 256 + threadIdx.x;
    if (i < n) out[i] = (__bf16)in[i];
}

// ---------------------------------------------------------------------------
// Fused dual GEMM on WMMA: out = agg @ W1^T + h @ W2^T + bias
// W1p/W2p are pre-packed bf16 [NCOLS,128]. Each block DMA-stages both weight
// images into LDS with global_load_async_to_lds_b128 (ASYNCcnt-tracked, no
// VGPR round trip), then each of the 4 waves computes a 16-row stripe across
// all column tiles. Out-of-range B columns are never stored, so no zero-fill.
// Layouts used here (contiguous k = 16*half + s, not the ISA table order):
//   A (16x32 bf16): lane m holds row m, K 0-15; lane m+16: K 16-31
//   B (32x16 bf16): lane = column (lane&15); half-wave selects K-half (16 each)
//   C/D (16x16 f32): vgpr v -> row v (lanes 0-15) / row v+8 (lanes 16-31)
// LDS offset of a flat pointer into the LDS aperture is its low 32 bits
// (ISA 10.2).
// ---------------------------------------------------------------------------
template<int NCOLS>
__global__ void gemm_wmma_kernel(const float*  __restrict__ A1,
                                 const float*  __restrict__ A2,
                                 const __bf16* __restrict__ W1p,
                                 const __bf16* __restrict__ W2p,
                                 const float*  __restrict__ bias,
                                 float* __restrict__ out)
{
    constexpr int NT = (NCOLS + 15) / 16;                 // column tiles
    constexpr unsigned MBYTES = NCOLS * DIM * sizeof(__bf16);
    __shared__ alignas(16) __bf16 sW[2 * NCOLS * DIM];
    __bf16* sW1 = sW;
    __bf16* sW2 = sW + NCOLS * DIM;

    // async DMA stage: global(bf16) -> LDS, 16B per lane per issue
    int tid = threadIdx.y * 32 + threadIdx.x;             // 0..127
    {
        unsigned ldsBase = (unsigned)(size_t)(void*)sW;   // low 32 bits = DS offset
        const char* g1 = (const char*)W1p;
        const char* g2 = (const char*)W2p;
        for (unsigned i = (unsigned)tid * 16u; i < MBYTES; i += 128u * 16u) {
            asm volatile("global_load_async_to_lds_b128 %0, %1, off"
                         :: "v"(ldsBase + i), "v"(g1 + i) : "memory");
            asm volatile("global_load_async_to_lds_b128 %0, %1, off"
                         :: "v"(ldsBase + MBYTES + i), "v"(g2 + i) : "memory");
        }
        asm volatile("s_wait_asynccnt 0" ::: "memory");
    }
    __syncthreads();

    int lane = threadIdx.x;                 // 0..31
    int r    = lane & 15;
    int half = lane >> 4;
    int row0 = (blockIdx.x * blockDim.y + threadIdx.y) * 16;
    if (row0 >= N_NODES) return;

    const float* a1row = A1 + (size_t)(row0 + r) * DIM;
    const float* a2row = A2 + (size_t)(row0 + r) * DIM;
    int cc = (r < (NCOLS & 15) || (NCOLS & 15) == 0) ? r : (NCOLS & 15) - 1;

    v8f acc[NT];
#pragma unroll
    for (int t = 0; t < NT; ++t) acc[t] = (v8f){};

#pragma unroll
    for (int ks = 0; ks < 4; ++ks) {
        const int k0 = ks * 32;
        if (ks < 3) {
            __builtin_prefetch(a1row + k0 + 32, 0, 1);
            __builtin_prefetch(a2row + k0 + 32, 0, 1);
        }
        // A fragments: aligned 32B vector loads + pack to bf16
        v8f a1lo = *(const v8f*)(a1row + k0 + half * 16);
        v8f a1hi = *(const v8f*)(a1row + k0 + half * 16 + 8);
        v8f a2lo = *(const v8f*)(a2row + k0 + half * 16);
        v8f a2hi = *(const v8f*)(a2row + k0 + half * 16 + 8);
        v16bf a1, a2;
#pragma unroll
        for (int j = 0; j < 8; ++j) {
            a1[j]     = (__bf16)a1lo[j];
            a1[8 + j] = (__bf16)a1hi[j];
            a2[j]     = (__bf16)a2lo[j];
            a2[8 + j] = (__bf16)a2hi[j];
        }
        // pass 1: agg @ W1^T (independent accumulators -> dependent WMMAs are
        // NT apart, hiding the WMMA->WMMA RAW hazard)
#pragma unroll
        for (int t = 0; t < NT; ++t) {
            int c = t * 16 + ((t == NT - 1) ? cc : r);
            v16bf b1 = *(const v16bf*)(sW1 + c * DIM + k0 + half * 16);
            acc[t] = __builtin_amdgcn_wmma_f32_16x16x32_bf16(
                false, a1, false, b1, (short)0, acc[t], false, false);
        }
        // pass 2: h @ W2^T
#pragma unroll
        for (int t = 0; t < NT; ++t) {
            int c = t * 16 + ((t == NT - 1) ? cc : r);
            v16bf b2 = *(const v16bf*)(sW2 + c * DIM + k0 + half * 16);
            acc[t] = __builtin_amdgcn_wmma_f32_16x16x32_bf16(
                false, a2, false, b2, (short)0, acc[t], false, false);
        }
    }

#pragma unroll
    for (int t = 0; t < NT; ++t) {
        int c = t * 16 + r;
        if (c < NCOLS) {
            float bv = bias[c];
#pragma unroll
            for (int v = 0; v < 8; ++v) {
                int row = row0 + half * 8 + v;
                out[(size_t)row * NCOLS + c] = acc[t][v] + bv;
            }
        }
    }
}

// ---------------------------------------------------------------------------
// BatchNorm statistics: one block per channel, biased variance (torch BN eval)
// ---------------------------------------------------------------------------
__global__ void bnstats_kernel(const float* __restrict__ h1,
                               float* __restrict__ mu, float* __restrict__ var)
{
    __shared__ float ssum[256];
    __shared__ float ssq[256];
    int c = blockIdx.x;
    float s = 0.f, q = 0.f;
    for (int n = threadIdx.x; n < N_NODES; n += 256) {
        float v = h1[(size_t)n * DIM + c];
        s += v;
        q = fmaf(v, v, q);
    }
    ssum[threadIdx.x] = s;
    ssq[threadIdx.x]  = q;
    __syncthreads();
    for (int off = 128; off > 0; off >>= 1) {
        if (threadIdx.x < off) {
            ssum[threadIdx.x] += ssum[threadIdx.x + off];
            ssq[threadIdx.x]  += ssq[threadIdx.x + off];
        }
        __syncthreads();
    }
    if (threadIdx.x == 0) {
        float m = ssum[0] * (1.0f / N_NODES);
        mu[c]  = m;
        var[c] = ssq[0] * (1.0f / N_NODES) - m * m;
    }
}

__global__ void bnapply_kernel(const float* __restrict__ h1,
                               const float* __restrict__ mu,
                               const float* __restrict__ var,
                               const float* __restrict__ gamma,
                               const float* __restrict__ beta,
                               float* __restrict__ h)
{
    int i = blockIdx.x * 256 + threadIdx.x;
    if (i >= N_NODES * DIM) return;
    int c = i & (DIM - 1);
    float inv = rsqrtf(var[c] + BN_EPS);
    float v = fmaf(gamma[c] * inv, h1[i] - mu[c], beta[c]);
    h[i] = fmaxf(v, 0.f);
}

// ---------------------------------------------------------------------------
extern "C" void kernel_launch(void* const* d_in, const int* in_sizes, int n_in,
                              void* d_out, int out_size, void* d_ws, size_t ws_size,
                              hipStream_t stream)
{
    const float* x      = (const float*)d_in[0];
    const int*   ei     = (const int*)  d_in[1];   // [2,E]
    const int*   labels = (const int*)  d_in[2];
    const float* conf   = (const float*)d_in[3];
    const float* W_enc  = (const float*)d_in[4];   // [3,C,K,D]
    const float* b_enc  = (const float*)d_in[5];   // [3,C,K]
    const float* W_dec  = (const float*)d_in[6];   // [3,C,D,K]
    const float* b_dec  = (const float*)d_in[7];   // [3,C,D]
    const float* Wl01   = (const float*)d_in[8];   // [2,D,D]
    const float* bl01   = (const float*)d_in[9];   // [2,D]
    const float* Wr01   = (const float*)d_in[10];  // [2,D,D]
    const float* Wl2    = (const float*)d_in[11];  // [C,D]
    const float* bl2    = (const float*)d_in[12];  // [C]
    const float* Wr2    = (const float*)d_in[13];  // [C,D]
    const float* gamma  = (const float*)d_in[14];  // [2,D]
    const float* beta   = (const float*)d_in[15];  // [2,D]
    float* out = (float*)d_out;

    // workspace carving (floats, then bf16 weight pack region)
    float* ws  = (float*)d_ws;
    size_t o = 0;
    float* enc  = ws + o; o += (size_t)N_NODES * 4;
    float* num  = ws + o; o += (size_t)N_NODES * DIM;   // then becomes agg
    float* den  = ws + o; o += (size_t)N_NODES;
    float* h1   = ws + o; o += (size_t)N_NODES * DIM;
    float* hbuf = ws + o; o += (size_t)N_NODES * DIM;
    float* mu   = ws + o; o += DIM;
    float* var  = ws + o; o += DIM;                      // o*4 is 16B aligned
    __bf16* wpL01 = (__bf16*)(ws + o);                   // [2,D,D]
    __bf16* wpR01 = wpL01 + (size_t)2 * DIM * DIM;       // [2,D,D]
    __bf16* wpL2  = wpR01 + (size_t)2 * DIM * DIM;       // [C,D]
    __bf16* wpR2  = wpL2  + (size_t)NCLASS * DIM;        // [C,D]
    (void)ws_size; (void)in_sizes; (void)n_in; (void)out_size;

    const int* srcI = ei;            // row 0
    const int* dstI = ei + N_EDGES;  // row 1

    const int encGrid = (N_NODES + 7) / 8;
    const int edgGrid = (N_EDGES + 7) / 8;
    const int elGrid  = (N_NODES * DIM + 255) / 256;
    const int rowTiles = N_NODES / 16;            // 3125, exact
    dim3 gblk(32, 4);                             // 4 waves per block
    dim3 ggrid((rowTiles + 3) / 4);

    // pack all GEMM weights to bf16 once
    pack_bf16_kernel<<<(2 * DIM * DIM + 255) / 256, 256, 0, stream>>>(Wl01, wpL01, 2 * DIM * DIM);
    pack_bf16_kernel<<<(2 * DIM * DIM + 255) / 256, 256, 0, stream>>>(Wr01, wpR01, 2 * DIM * DIM);
    pack_bf16_kernel<<<(NCLASS * DIM + 255) / 256, 256, 0, stream>>>(Wl2, wpL2, NCLASS * DIM);
    pack_bf16_kernel<<<(NCLASS * DIM + 255) / 256, 256, 0, stream>>>(Wr2, wpR2, NCLASS * DIM);

    const float* hin = x;
    for (int l = 0; l < 2; ++l) {
        hipMemsetAsync(num, 0, sizeof(float) * (size_t)N_NODES * DIM, stream);
        hipMemsetAsync(den, 0, sizeof(float) * (size_t)N_NODES, stream);
        enc_kernel<<<encGrid, 256, 0, stream>>>(
            hin, labels,
            W_enc + (size_t)l * NCLASS * KB * DIM,
            b_enc + (size_t)l * NCLASS * KB, enc);
        edge_kernel<<<edgGrid, 256, 0, stream>>>(
            srcI, dstI, conf, labels, enc,
            W_dec + (size_t)l * NCLASS * DIM * KB,
            b_dec + (size_t)l * NCLASS * DIM, num, den);
        finalize_kernel<<<elGrid, 256, 0, stream>>>(num, den);
        gemm_wmma_kernel<DIM><<<ggrid, gblk, 0, stream>>>(
            num, hin,
            wpL01 + (size_t)l * DIM * DIM,
            wpR01 + (size_t)l * DIM * DIM,
            bl01 + (size_t)l * DIM, h1);
        bnstats_kernel<<<DIM, 256, 0, stream>>>(h1, mu, var);
        bnapply_kernel<<<elGrid, 256, 0, stream>>>(
            h1, mu, var, gamma + (size_t)l * DIM, beta + (size_t)l * DIM, hbuf);
        hin = hbuf;
    }

    // final layer: out[N, 40]
    hipMemsetAsync(num, 0, sizeof(float) * (size_t)N_NODES * DIM, stream);
    hipMemsetAsync(den, 0, sizeof(float) * (size_t)N_NODES, stream);
    enc_kernel<<<encGrid, 256, 0, stream>>>(
        hin, labels,
        W_enc + (size_t)2 * NCLASS * KB * DIM,
        b_enc + (size_t)2 * NCLASS * KB, enc);
    edge_kernel<<<edgGrid, 256, 0, stream>>>(
        srcI, dstI, conf, labels, enc,
        W_dec + (size_t)2 * NCLASS * DIM * KB,
        b_dec + (size_t)2 * NCLASS * DIM, num, den);
    finalize_kernel<<<elGrid, 256, 0, stream>>>(num, den);
    gemm_wmma_kernel<NCLASS><<<ggrid, gblk, 0, stream>>>(
        num, hin, wpL2, wpR2, bl2, out);
}
